// DbrxBlock_35957466202273
// MI455X (gfx1250) — matrix operand order repaired and checked by reading, not compiled, not run on hardware
//
#include <hip/hip_runtime.h>

#define SQ    2048
#define DIM   1024
#define NH    16
#define NKV   4
#define HDIM  64
#define NE    16
#define NI    4096
#define NTOP  4
#define CLIPV 8.0f
#define ROWCAP (NTOP*SQ + NE*128)   /* 10240 gathered rows (128-aligned per expert) */
#define ROWTILES (ROWCAP/128)       /* 80 */

typedef float    v8f  __attribute__((ext_vector_type(8)));
typedef _Float16 v16h __attribute__((ext_vector_type(16)));
typedef _Float16 v8h  __attribute__((ext_vector_type(8)));

union F16frag { v16h v; v8h h[2]; };

// Async global->LDS 16-byte copy (CDNA5 GLOBAL_LOAD_ASYNC_TO_LDS_B128, ASYNCcnt).
__device__ __forceinline__ void async_copy_b128(const void* gptr, void* lptr) {
  unsigned lds = (unsigned)(uintptr_t)lptr;   // low 32 bits of flat LDS address = LDS offset
  asm volatile("global_load_async_to_lds_b128 %0, %1, off"
               :: "v"(lds), "v"(gptr) : "memory");
}
__device__ __forceinline__ void async_wait() {
  asm volatile("s_wait_asynccnt 0" ::: "memory");
}

// ---------------- LayerNorm -> f16 ----------------
__global__ void ln_kernel(const float* __restrict__ in, const float* __restrict__ w,
                          const float* __restrict__ b, _Float16* __restrict__ outh) {
  int t = blockIdx.x, tid = threadIdx.x;            // 256 threads, D=1024
  __shared__ float red[256];
  const float* x = in + (size_t)t * DIM;
  float vals[4]; float s = 0.f;
  #pragma unroll
  for (int i = 0; i < 4; i++) { vals[i] = x[tid + 256*i]; s += vals[i]; }
  red[tid] = s; __syncthreads();
  for (int st = 128; st > 0; st >>= 1) { if (tid < st) red[tid] += red[tid+st]; __syncthreads(); }
  float mean = red[0] * (1.f/DIM);
  __syncthreads();
  float v = 0.f;
  #pragma unroll
  for (int i = 0; i < 4; i++) { float d = vals[i]-mean; v += d*d; }
  red[tid] = v; __syncthreads();
  for (int st = 128; st > 0; st >>= 1) { if (tid < st) red[tid] += red[tid+st]; __syncthreads(); }
  float rstd = rsqrtf(red[0] * (1.f/DIM) + 1e-5f);
  #pragma unroll
  for (int i = 0; i < 4; i++) {
    int d = tid + 256*i;
    outh[(size_t)t*DIM + d] = (_Float16)((vals[i]-mean)*rstd*w[d] + b[d]);
  }
}

// ---------------- Generic 128x128 WMMA GEMM: C = A(f16) * B(f32->f16)^T ----------------
// MODE 0: clip(+-CLIPV), store f32.  MODE 1: + resid, store f32.
template <int MODE>
__global__ __launch_bounds__(256) void gemm128(const _Float16* __restrict__ A, int lda,
                                               const float* __restrict__ B, int ldb,
                                               float* __restrict__ C, int ldc, int K,
                                               const float* __restrict__ resid) {
  __shared__ __align__(16) _Float16 As[128][40];
  __shared__ __align__(16) _Float16 Bs[128][40];
  int tid = threadIdx.x, lane = tid & 31, wave = tid >> 5;
  int wm = wave & 3, wn = wave >> 2;
  int bx = blockIdx.x, by = blockIdx.y;
  v8f acc[2][4] = {};
  for (int kc = 0; kc < K; kc += 32) {
    if (kc + 32 < K)                                    // prefetch next weight chunk
      __builtin_prefetch(&B[(size_t)(bx*128 + (tid >> 1)) * ldb + kc + 32 + (tid & 1)*16], 0, 0);
    #pragma unroll
    for (int v = tid; v < 512; v += 256) {              // A: 128x32 f16, async -> LDS
      int row = v >> 2, kv = (v & 3) * 8;
      async_copy_b128(&A[(size_t)(by*128 + row) * lda + kc + kv], &As[row][kv]);
    }
    for (int v = tid; v < 1024; v += 256) {             // B: 128x32 f32 -> f16
      int row = v >> 3, kq = (v & 7) * 4;
      const float4 f = *(const float4*)&B[(size_t)(bx*128 + row) * ldb + kc + kq];
      _Float16* d = &Bs[row][kq];
      d[0]=(_Float16)f.x; d[1]=(_Float16)f.y; d[2]=(_Float16)f.z; d[3]=(_Float16)f.w;
    }
    async_wait();
    __syncthreads();
    F16frag af[2], bf[4];
    int ar = wm*32 + (lane & 15), akb = (lane < 16) ? 0 : 16;
    #pragma unroll
    for (int i = 0; i < 2; i++) {
      af[i].h[0] = *(const v8h*)&As[ar + i*16][akb];
      af[i].h[1] = *(const v8h*)&As[ar + i*16][akb + 8];
    }
    int bc = wn*64 + (lane & 15), bkb = (lane < 16) ? 0 : 16;
    #pragma unroll
    for (int j = 0; j < 4; j++) {
      bf[j].h[0] = *(const v8h*)&Bs[bc + j*16][bkb];
      bf[j].h[1] = *(const v8h*)&Bs[bc + j*16][bkb + 8];
    }
    #pragma unroll
    for (int i = 0; i < 2; i++)
      #pragma unroll
      for (int j = 0; j < 4; j++)
        acc[i][j] = __builtin_amdgcn_wmma_f32_16x16x32_f16(false, af[i].v, false, bf[j].v,
                                                           (short)0, acc[i][j], false, false);
    __syncthreads();
  }
  int rbase = by*128 + wm*32 + ((lane < 16) ? 0 : 8);
  int cbase = bx*128 + wn*64 + (lane & 15);
  #pragma unroll
  for (int i = 0; i < 2; i++)
    #pragma unroll
    for (int j = 0; j < 4; j++)
      #pragma unroll
      for (int p = 0; p < 8; p++) {
        int r = rbase + i*16 + p, c = cbase + j*16;
        float v = acc[i][j][p];
        if (MODE == 0) v = fminf(fmaxf(v, -CLIPV), CLIPV);
        if (MODE == 1) v += resid[(size_t)r*ldc + c];
        C[(size_t)r*ldc + c] = v;
      }
}

// ---------------- RoPE on q (16 heads) and k (4 heads), in place ----------------
__global__ void rope_kernel(float* __restrict__ qkv, const int* __restrict__ pos) {
  int t = blockIdx.x;
  float p = (float)pos[t];
  float* row = qkv + (size_t)t * 1536;
  for (int i = threadIdx.x; i < 20*32; i += blockDim.x) {
    int hh = i >> 5, fi = i & 31;
    float* base = row + ((hh < NH) ? hh*HDIM : NH*HDIM + (hh-NH)*HDIM);
    float inv = powf(500000.0f, -(float)fi / 32.0f);
    float ang = p * inv;
    float c = cosf(ang), s = sinf(ang);
    float x1 = base[fi], x2 = base[fi + 32];
    base[fi]      = x1*c - x2*s;
    base[fi + 32] = x2*c + x1*s;
  }
}

// ---------------- WMMA flash attention ----------------
// One block per (head, 128-query tile). 8 waves x 16 query rows. Key tiles of 32.
// K staged row-major (B-frag-ready for Q*K^T), V staged transposed (B-frag-ready for P*V).
// P is re-shaped C-layout -> A-layout through a per-wave LDS tile.
__global__ __launch_bounds__(256) void flash_attn_kernel(const float* __restrict__ qkv,
                                                         _Float16* __restrict__ attnh) {
  __shared__ __align__(16) _Float16 Qs[128][72];   // 128 q rows x 64 hd
  __shared__ __align__(16) _Float16 Ks[32][72];    // 32 keys x 64 hd
  __shared__ __align__(16) _Float16 Vt[64][40];    // 64 hd x 32 keys (transposed)
  __shared__ __align__(16) _Float16 Ps[8][16][40]; // per-wave P tile 16x32

  int h = blockIdx.x, qb = blockIdx.y * 128;
  int tid = threadIdx.x, lane = tid & 31, w = tid >> 5;
  int kvh = h >> 2;
  int lo = (lane < 16);
  int ln16 = lane & 15;

  // stage Q (scaled by HD^-0.5) : 128x64 floats = 2048 float4
  for (int v = tid; v < 2048; v += 256) {
    int row = v >> 4, d0 = (v & 15) * 4;
    const float4 f = *(const float4*)&qkv[(size_t)(qb + row) * 1536 + h*HDIM + d0];
    _Float16* d = &Qs[row][d0];
    d[0]=(_Float16)(f.x*0.125f); d[1]=(_Float16)(f.y*0.125f);
    d[2]=(_Float16)(f.z*0.125f); d[3]=(_Float16)(f.w*0.125f);
  }
  __syncthreads();

  // per-wave Q A-fragments (16 rows x 64 hd = 2 fragments), kept in registers
  F16frag aq[2];
  {
    int qr = w*16 + ln16, akb = lo ? 0 : 16;
    #pragma unroll
    for (int c = 0; c < 2; c++) {
      aq[c].h[0] = *(const v8h*)&Qs[qr][c*32 + akb];
      aq[c].h[1] = *(const v8h*)&Qs[qr][c*32 + akb + 8];
    }
  }

  int rw0 = qb + w*16;                 // first query row of this wave
  float mrow[8], lrow[8];
  #pragma unroll
  for (int p = 0; p < 8; p++) { mrow[p] = -3e38f; lrow[p] = 0.f; }
  v8f o[4] = {};

  int ntiles = qb/32 + 4;              // keys 0 .. qb+127
  for (int kt = 0; kt < ntiles; kt++) {
    __syncthreads();
    // stage K tile (row-major f16) and V tile (transposed f16)
    for (int v = tid; v < 512; v += 256) {
      int row = v >> 4, d0 = (v & 15) * 4;
      const float4 f = *(const float4*)&qkv[(size_t)(kt*32 + row) * 1536 + NH*HDIM + kvh*HDIM + d0];
      _Float16* d = &Ks[row][d0];
      d[0]=(_Float16)f.x; d[1]=(_Float16)f.y; d[2]=(_Float16)f.z; d[3]=(_Float16)f.w;
    }
    for (int v = tid; v < 512; v += 256) {
      int row = v >> 4, d0 = (v & 15) * 4;
      const float4 f = *(const float4*)&qkv[(size_t)(kt*32 + row) * 1536 + (NH+NKV)*HDIM + kvh*HDIM + d0];
      Vt[d0+0][row]=(_Float16)f.x; Vt[d0+1][row]=(_Float16)f.y;
      Vt[d0+2][row]=(_Float16)f.z; Vt[d0+3][row]=(_Float16)f.w;
    }
    __syncthreads();

    if (kt*32 > rw0 + 15) continue;    // tile fully above the causal frontier (wave-uniform)

    // S = Q K^T : two 16x16 key groups, K accumulated over hd chunks
    v8f s[2] = {};
    #pragma unroll
    for (int j = 0; j < 2; j++) {
      F16frag bk0, bk1;
      int key = j*16 + ln16;
      int off0 = lo ? 0 : 16, off1 = 32 + (lo ? 0 : 16);
      bk0.h[0] = *(const v8h*)&Ks[key][off0]; bk0.h[1] = *(const v8h*)&Ks[key][off0 + 8];
      bk1.h[0] = *(const v8h*)&Ks[key][off1]; bk1.h[1] = *(const v8h*)&Ks[key][off1 + 8];
      s[j] = __builtin_amdgcn_wmma_f32_16x16x32_f16(false, aq[0].v, false, bk0.v,
                                                    (short)0, s[j], false, false);
      s[j] = __builtin_amdgcn_wmma_f32_16x16x32_f16(false, aq[1].v, false, bk1.v,
                                                    (short)0, s[j], false, false);
    }
    // causal mask (exact, element-wise in C layout)
    int rowb = rw0 + (lo ? 0 : 8);
    #pragma unroll
    for (int j = 0; j < 2; j++) {
      int col = kt*32 + j*16 + ln16;
      #pragma unroll
      for (int p = 0; p < 8; p++)
        if (col > rowb + p) s[j][p] = -1e30f;
    }
    // online softmax: row stats via 16-lane butterflies (rows indexed by p per half)
    float corr[8];
    v8f p0, p1;
    #pragma unroll
    for (int p = 0; p < 8; p++) {
      float v = fmaxf(s[0][p], s[1][p]);
      #pragma unroll
      for (int off = 8; off >= 1; off >>= 1) v = fmaxf(v, __shfl_xor(v, off));
      float mn = fmaxf(mrow[p], v);
      p0[p] = __expf(s[0][p] - mn);
      p1[p] = __expf(s[1][p] - mn);
      float rs = p0[p] + p1[p];
      #pragma unroll
      for (int off = 8; off >= 1; off >>= 1) rs += __shfl_xor(rs, off);
      corr[p] = __expf(mrow[p] - mn);
      lrow[p] = lrow[p]*corr[p] + rs;
      mrow[p] = mn;
    }
    #pragma unroll
    for (int j = 0; j < 4; j++)
      #pragma unroll
      for (int p = 0; p < 8; p++) o[j][p] *= corr[p];

    // P: C-layout -> per-wave LDS -> A-layout fragment (same-wave LDS ops are in order)
    int rloc = lo ? 0 : 8;
    #pragma unroll
    for (int p = 0; p < 8; p++) {
      Ps[w][rloc + p][ln16]      = (_Float16)p0[p];
      Ps[w][rloc + p][ln16 + 16] = (_Float16)p1[p];
    }
    F16frag pf;
    {
      int pkb = lo ? 0 : 16;
      pf.h[0] = *(const v8h*)&Ps[w][ln16][pkb];
      pf.h[1] = *(const v8h*)&Ps[w][ln16][pkb + 8];
    }
    // O += P V : 4 hd groups of 16
    #pragma unroll
    for (int j = 0; j < 4; j++) {
      F16frag bv;
      int hd = j*16 + ln16, sel = lo ? 0 : 16;
      bv.h[0] = *(const v8h*)&Vt[hd][sel];
      bv.h[1] = *(const v8h*)&Vt[hd][sel + 8];
      o[j] = __builtin_amdgcn_wmma_f32_16x16x32_f16(false, pf.v, false, bv.v,
                                                    (short)0, o[j], false, false);
    }
  }

  // normalize and store f16
  int rowb = rw0 + (lo ? 0 : 8);
  #pragma unroll
  for (int p = 0; p < 8; p++) {
    float inv = __builtin_amdgcn_rcpf(lrow[p]);
    #pragma unroll
    for (int j = 0; j < 4; j++)
      attnh[(size_t)(rowb + p) * DIM + h*HDIM + j*16 + ln16] = (_Float16)(o[j][p] * inv);
  }
}

// ---------------- Router: softmax over 16 experts, top-4, slot assignment ----------------
__global__ void router_kernel(const _Float16* __restrict__ xh2, const float* __restrict__ rw,
                              float* __restrict__ gates, int* __restrict__ eidx,
                              int* __restrict__ slotin, int* __restrict__ counts) {
  int t = blockIdx.x, lane = threadIdx.x;               // 32 threads
  float logit = -1e30f;
  if (lane < NE) {
    const _Float16* x = xh2 + (size_t)t * DIM;
    const float* w = rw + (size_t)lane * DIM;
    float s = 0.f;
    for (int d = 0; d < DIM; d += 2) s += (float)x[d]*w[d] + (float)x[d+1]*w[d+1];
    logit = s;
  }
  float mx = logit;
  #pragma unroll
  for (int off = 8; off >= 1; off >>= 1) mx = fmaxf(mx, __shfl_xor(mx, off));
  float p = (lane < NE) ? __expf(logit - mx) : 0.f;
  float dsum = p;
  #pragma unroll
  for (int off = 8; off >= 1; off >>= 1) dsum += __shfl_xor(dsum, off);
  float prob = (lane < NE) ? p / dsum : -1.f;
  float pv = prob;
  float g[NTOP]; int ei[NTOP];
  #pragma unroll
  for (int k = 0; k < NTOP; k++) {
    float bv = pv; int bi = lane;
    #pragma unroll
    for (int off = 8; off >= 1; off >>= 1) {
      float ov = __shfl_xor(bv, off); int oi = __shfl_xor(bi, off);
      if (ov > bv || (ov == bv && oi < bi)) { bv = ov; bi = oi; }
    }
    g[k] = bv; ei[k] = bi;
    if (lane == bi) pv = -1.f;
  }
  if (lane == 0) {
    float norm = g[0] + g[1] + g[2] + g[3];
    #pragma unroll
    for (int k = 0; k < NTOP; k++) {
      int slot = atomicAdd(&counts[ei[k]], 1);
      gates[t*NTOP + k]  = g[k] / norm;
      eidx[t*NTOP + k]   = ei[k];
      slotin[t*NTOP + k] = slot;
    }
  }
}

__global__ void scan_kernel(const int* __restrict__ counts, int* __restrict__ offsets) {
  if (threadIdx.x == 0 && blockIdx.x == 0) {
    int off = 0;
    for (int e = 0; e < NE; e++) { offsets[e] = off; off += (counts[e] + 127) & ~127; }
  }
}

__global__ void build_rows_kernel(const int* __restrict__ eidx, const int* __restrict__ slotin,
                                  const float* __restrict__ gates, const int* __restrict__ offsets,
                                  int* __restrict__ rowtok, float* __restrict__ rowgate,
                                  int* __restrict__ rowof) {
  int i = blockIdx.x * blockDim.x + threadIdx.x;
  if (i >= SQ * NTOP) return;
  int t = i >> 2;
  int r = offsets[eidx[i]] + slotin[i];
  rowtok[r]  = t;
  rowgate[r] = gates[i];
  rowof[i]   = r;
}

__device__ __forceinline__ int find_expert(int r0, const int* counts, const int* offsets) {
  for (int e = 0; e < NE; e++) {
    int pad = (counts[e] + 127) & ~127;
    if (r0 >= offsets[e] && r0 < offsets[e] + pad) return e;
  }
  return -1;
}

// ---------------- MoE stage 1: act = silu(x*W1^T) * (x*W2^T), gathered rows ----------------
__global__ __launch_bounds__(256) void moe1_kernel(const _Float16* __restrict__ xh2,
                                                   const float* __restrict__ wsAll,
                                                   const int* __restrict__ rowtok,
                                                   const int* __restrict__ counts,
                                                   const int* __restrict__ offsets,
                                                   _Float16* __restrict__ act) {
  int r0 = blockIdx.y * 128;
  int total = offsets[NE-1] + ((counts[NE-1] + 127) & ~127);
  if (r0 >= total) return;
  int e = find_expert(r0, counts, offsets);
  const float* wsE = wsAll + (size_t)e * (2*NI) * DIM;
  int n0 = blockIdx.x * 64;

  __shared__ __align__(16) _Float16 As[128][40];
  __shared__ __align__(16) _Float16 B1s[64][40];
  __shared__ __align__(16) _Float16 B2s[64][40];
  int tid = threadIdx.x, lane = tid & 31, wave = tid >> 5;
  int wm = wave & 3, wn = wave >> 2;
  v8f acc1[2][2] = {}, acc2[2][2] = {};

  for (int kc = 0; kc < DIM; kc += 32) {
    if (kc + 32 < DIM) {
      __builtin_prefetch(&wsE[(size_t)(n0 + (tid >> 2)) * DIM + kc + 32 + (tid & 3)*8], 0, 0);
      __builtin_prefetch(&wsE[(size_t)(NI + n0 + (tid >> 2)) * DIM + kc + 32 + (tid & 3)*8], 0, 0);
    }
    for (int v = tid; v < 512; v += 256) {              // gathered A
      int row = v >> 2, kv = (v & 3) * 8;
      int tok = rowtok[r0 + row];
      v8h val = {};
      if (tok >= 0) val = *(const v8h*)&xh2[(size_t)tok * DIM + kc + kv];
      *(v8h*)&As[row][kv] = val;
    }
    for (int v = tid; v < 1024; v += 256) {             // W1 / W2 tiles
      int hsel = v >> 9, vv = v & 511;
      int row = vv >> 3, kq = (vv & 7) * 4;
      size_t srow = hsel ? (size_t)(NI + n0 + row) : (size_t)(n0 + row);
      const float4 f = *(const float4*)&wsE[srow * DIM + kc + kq];
      _Float16* d = hsel ? &B2s[row][kq] : &B1s[row][kq];
      d[0]=(_Float16)f.x; d[1]=(_Float16)f.y; d[2]=(_Float16)f.z; d[3]=(_Float16)f.w;
    }
    __syncthreads();
    F16frag af[2], b1[2], b2[2];
    int ar = wm*32 + (lane & 15), akb = (lane < 16) ? 0 : 16;
    #pragma unroll
    for (int i = 0; i < 2; i++) {
      af[i].h[0] = *(const v8h*)&As[ar + i*16][akb];
      af[i].h[1] = *(const v8h*)&As[ar + i*16][akb + 8];
    }
    int bc = wn*32 + (lane & 15), bkb = (lane < 16) ? 0 : 16;
    #pragma unroll
    for (int j = 0; j < 2; j++) {
      b1[j].h[0] = *(const v8h*)&B1s[bc + j*16][bkb];
      b1[j].h[1] = *(const v8h*)&B1s[bc + j*16][bkb + 8];
      b2[j].h[0] = *(const v8h*)&B2s[bc + j*16][bkb];
      b2[j].h[1] = *(const v8h*)&B2s[bc + j*16][bkb + 8];
    }
    #pragma unroll
    for (int i = 0; i < 2; i++)
      #pragma unroll
      for (int j = 0; j < 2; j++) {
        acc1[i][j] = __builtin_amdgcn_wmma_f32_16x16x32_f16(false, af[i].v, false, b1[j].v,
                                                            (short)0, acc1[i][j], false, false);
        acc2[i][j] = __builtin_amdgcn_wmma_f32_16x16x32_f16(false, af[i].v, false, b2[j].v,
                                                            (short)0, acc2[i][j], false, false);
      }
    __syncthreads();
  }
  int rbase = r0 + wm*32 + ((lane < 16) ? 0 : 8);
  int cbase = n0 + wn*32 + (lane & 15);
  #pragma unroll
  for (int i = 0; i < 2; i++)
    #pragma unroll
    for (int j = 0; j < 2; j++)
      #pragma unroll
      for (int p = 0; p < 8; p++) {
        float h1 = acc1[i][j][p], h2 = acc2[i][j][p];
        float a = h1 * __builtin_amdgcn_rcpf(1.f + __expf(-h1)) * h2;   // silu(h1)*h2
        act[(size_t)(rbase + i*16 + p) * NI + cbase + j*16] = (_Float16)a;
      }
}

// ---------------- MoE stage 2: y = gate * (act @ w2s[e]^T) -> ybuf ----------------
__global__ __launch_bounds__(256) void moe2_kernel(const _Float16* __restrict__ act,
                                                   const float* __restrict__ w2sAll,
                                                   const float* __restrict__ rowgate,
                                                   const int* __restrict__ counts,
                                                   const int* __restrict__ offsets,
                                                   float* __restrict__ ybuf) {
  int r0 = blockIdx.y * 128;
  int total = offsets[NE-1] + ((counts[NE-1] + 127) & ~127);
  if (r0 >= total) return;
  int e = find_expert(r0, counts, offsets);
  const float* B = w2sAll + (size_t)e * DIM * NI;
  int bx = blockIdx.x;

  __shared__ __align__(16) _Float16 As[128][40];
  __shared__ __align__(16) _Float16 Bs[128][40];
  int tid = threadIdx.x, lane = tid & 31, wave = tid >> 5;
  int wm = wave & 3, wn = wave >> 2;
  v8f acc[2][4] = {};
  for (int kc = 0; kc < NI; kc += 32) {
    if (kc + 32 < NI)
      __builtin_prefetch(&B[(size_t)(bx*128 + (tid >> 1)) * NI + kc + 32 + (tid & 1)*16], 0, 0);
    #pragma unroll
    for (int v = tid; v < 512; v += 256) {              // A (f16 act rows), async -> LDS
      int row = v >> 2, kv = (v & 3) * 8;
      async_copy_b128(&act[(size_t)(r0 + row) * NI + kc + kv], &As[row][kv]);
    }
    for (int v = tid; v < 1024; v += 256) {
      int row = v >> 3, kq = (v & 7) * 4;
      const float4 f = *(const float4*)&B[(size_t)(bx*128 + row) * NI + kc + kq];
      _Float16* d = &Bs[row][kq];
      d[0]=(_Float16)f.x; d[1]=(_Float16)f.y; d[2]=(_Float16)f.z; d[3]=(_Float16)f.w;
    }
    async_wait();
    __syncthreads();
    F16frag af[2], bf[4];
    int ar = wm*32 + (lane & 15), akb = (lane < 16) ? 0 : 16;
    #pragma unroll
    for (int i = 0; i < 2; i++) {
      af[i].h[0] = *(const v8h*)&As[ar + i*16][akb];
      af[i].h[1] = *(const v8h*)&As[ar + i*16][akb + 8];
    }
    int bc = wn*64 + (lane & 15), bkb = (lane < 16) ? 0 : 16;
    #pragma unroll
    for (int j = 0; j < 4; j++) {
      bf[j].h[0] = *(const v8h*)&Bs[bc + j*16][bkb];
      bf[j].h[1] = *(const v8h*)&Bs[bc + j*16][bkb + 8];
    }
    #pragma unroll
    for (int i = 0; i < 2; i++)
      #pragma unroll
      for (int j = 0; j < 4; j++)
        acc[i][j] = __builtin_amdgcn_wmma_f32_16x16x32_f16(false, af[i].v, false, bf[j].v,
                                                           (short)0, acc[i][j], false, false);
    __syncthreads();
  }
  int rbase = r0 + wm*32 + ((lane < 16) ? 0 : 8);
  int cbase = bx*128 + wn*64 + (lane & 15);
  #pragma unroll
  for (int i = 0; i < 2; i++)
    #pragma unroll
    for (int p = 0; p < 8; p++) {
      int r = rbase + i*16 + p;
      float g = rowgate[r];
      #pragma unroll
      for (int j = 0; j < 4; j++)
        ybuf[(size_t)r * DIM + cbase + j*16] = acc[i][j][p] * g;
    }
}

// ---------------- Final gather: out = hidden + sum_k ybuf[rowof[t][k]] ----------------
__global__ void final_kernel(const float* __restrict__ hidden, const float* __restrict__ ybuf,
                             const int* __restrict__ rowof, float* __restrict__ out) {
  int t = blockIdx.x;
  int d = blockIdx.y * 256 + threadIdx.x;
  float v = hidden[(size_t)t*DIM + d];
  #pragma unroll
  for (int k = 0; k < NTOP; k++)
    v += ybuf[(size_t)rowof[t*NTOP + k] * DIM + d];
  out[(size_t)t*DIM + d] = v;
}

extern "C" void kernel_launch(void* const* d_in, const int* in_sizes, int n_in,
                              void* d_out, int out_size, void* d_ws, size_t ws_size,
                              hipStream_t stream) {
  (void)in_sizes; (void)n_in; (void)out_size; (void)ws_size;
  const float* hs    = (const float*)d_in[0];
  const int*   pos   = (const int*)  d_in[1];
  const float* ln1w  = (const float*)d_in[2];
  const float* ln1b  = (const float*)d_in[3];
  const float* ln2w  = (const float*)d_in[4];
  const float* ln2b  = (const float*)d_in[5];
  const float* wqkv  = (const float*)d_in[6];
  const float* wout  = (const float*)d_in[7];
  const float* rw    = (const float*)d_in[8];
  const float* wsAll = (const float*)d_in[9];
  const float* w2s   = (const float*)d_in[10];
  float* out = (float*)d_out;

  char* p = (char*)d_ws;
  auto grab = [&](size_t bytes) { void* r = (void*)p; p += (bytes + 255) & ~(size_t)255; return r; };
  _Float16* xh1   = (_Float16*)grab((size_t)SQ*DIM*2);
  _Float16* xh2   = (_Float16*)grab((size_t)SQ*DIM*2);
  float*    qkvb  = (float*)   grab((size_t)SQ*1536*4);
  _Float16* attnh = (_Float16*)grab((size_t)SQ*DIM*2);
  float*    hidden= (float*)   grab((size_t)SQ*DIM*4);
  _Float16* acth  = (_Float16*)grab((size_t)ROWCAP*NI*2);
  float*    ybuf  = (float*)   grab((size_t)ROWCAP*DIM*4);
  float*    gates = (float*)   grab((size_t)SQ*NTOP*4);
  int*      eidx  = (int*)     grab((size_t)SQ*NTOP*4);
  int*      slotin= (int*)     grab((size_t)SQ*NTOP*4);
  int*      rowof = (int*)     grab((size_t)SQ*NTOP*4);
  int*      rowtok= (int*)     grab((size_t)ROWCAP*4);
  float*    rowgate=(float*)   grab((size_t)ROWCAP*4);
  int*      counts= (int*)     grab(NE*4);
  int*      offsets=(int*)     grab(NE*4);

  // 1. LN1 -> f16
  ln_kernel<<<SQ, 256, 0, stream>>>(hs, ln1w, ln1b, xh1);
  // 2. QKV GEMM + clip (M=2048, N=1536, K=1024)
  gemm128<0><<<dim3(1536/128, SQ/128), 256, 0, stream>>>(xh1, DIM, wqkv, DIM, qkvb, 1536, DIM, nullptr);
  // 3. RoPE
  rope_kernel<<<SQ, 256, 0, stream>>>(qkvb, pos);
  // 4. Causal GQA flash attention (WMMA)
  flash_attn_kernel<<<dim3(NH, SQ/128), 256, 0, stream>>>(qkvb, attnh);
  // 5. Output projection + residual
  gemm128<1><<<dim3(DIM/128, SQ/128), 256, 0, stream>>>(attnh, DIM, wout, DIM, hidden, DIM, DIM, hs);
  // 6. LN2 -> f16
  ln_kernel<<<SQ, 256, 0, stream>>>(hidden, ln2w, ln2b, xh2);
  // 7. Routing
  hipMemsetAsync(counts, 0, NE*4, stream);
  hipMemsetAsync(rowtok, 0xFF, (size_t)ROWCAP*4, stream);
  router_kernel<<<SQ, 32, 0, stream>>>(xh2, rw, gates, eidx, slotin, counts);
  scan_kernel<<<1, 32, 0, stream>>>(counts, offsets);
  build_rows_kernel<<<(SQ*NTOP + 255)/256, 256, 0, stream>>>(eidx, slotin, gates, offsets,
                                                             rowtok, rowgate, rowof);
  // 8. MoE stage 1 (gathered, fused SwiGLU)
  moe1_kernel<<<dim3(NI/64, ROWTILES), 256, 0, stream>>>(xh2, wsAll, rowtok, counts, offsets, acth);
  // 9. MoE stage 2 (gate-scaled)
  moe2_kernel<<<dim3(DIM/128, ROWTILES), 256, 0, stream>>>(acth, w2s, rowgate, counts, offsets, ybuf);
  // 10. Final gather + residual
  final_kernel<<<dim3(SQ, DIM/256), 256, 0, stream>>>(hidden, ybuf, rowof, out);
}
